// VoxelBackBone8x_83253646065898
// MI455X (gfx1250) — matrix operand order repaired and checked by reading, not compiled, not run on hardware
//
#include <hip/hip_runtime.h>
#include <hip/hip_bf16.h>

// ---------------------------------------------------------------------------
// MI455X (gfx1250) sparse-conv backbone.
// Core op: float4 gather -> v_wmma_f32_16x16x32_bf16 -> global_atomic_add_f32.
// Weights are pre-swizzled once per layer into per-lane bf16 B fragments so the
// hot loop issues one aligned 32B load per fragment (no VALU conversion).
// Wave32; one wave = 16 rulebook rows x full Cout for one kernel offset.
// ---------------------------------------------------------------------------

typedef __bf16 bf16_t;
typedef bf16_t         v16bf __attribute__((ext_vector_type(16)));
typedef unsigned short v16us __attribute__((ext_vector_type(16)));
typedef float           v8f  __attribute__((ext_vector_type(8)));

__device__ __forceinline__ unsigned short f2bf_bits(float x) {
    // round-to-nearest-even fp32 -> bf16
    unsigned u = __float_as_uint(x);
    unsigned r = u + 0x7fffu + ((u >> 16) & 1u);
    return (unsigned short)(r >> 16);
}

union BFrag { v16us u; v16bf b; };

// ---------------------------------------------------------------------------
// mean-VFE: x[i,c] = sum_j voxels[i,j,c] / max(num_points[i],1); zero sentinel row
// ---------------------------------------------------------------------------
__global__ void vfe_kernel(const float* __restrict__ voxels,
                           const int* __restrict__ nump,
                           float* __restrict__ x0, int N) {
    int idx = blockIdx.x * blockDim.x + threadIdx.x;
    int total = (N + 1) * 4;
    if (idx >= total) return;
    int i = idx >> 2, c = idx & 3;
    float y = 0.f;
    if (i < N) {
        float s = 0.f;
        #pragma unroll
        for (int j = 0; j < 5; ++j) s += voxels[(i * 5 + j) * 4 + c];
        y = s / fmaxf((float)nump[i], 1.f);
    }
    x0[idx] = y;
}

// ---------------------------------------------------------------------------
// Weight pre-swizzle: (K, CIN, COUT) fp32 -> bf16 WMMA B fragments.
// Fragment index ((koff*NT + nt)*KB + kb); each fragment = 32 lanes x 16 halves.
// ISA 16-bit B layout: lane holds column n = lane&15, k = (h<8?h:h+8) + (lane>>4)*8.
// ---------------------------------------------------------------------------
template<int CIN, int COUT>
__global__ void preswz_kernel(const float* __restrict__ w,
                              unsigned short* __restrict__ swz, int K) {
    constexpr int KB = (CIN + 31) / 32;
    constexpr int NT = COUT / 16;
    int t = blockIdx.x * blockDim.x + threadIdx.x;
    int total = K * NT * KB * 32;
    if (t >= total) return;
    int lane = t & 31;
    int frag = t >> 5;
    int kb   = frag % KB;
    int rem  = frag / KB;
    int nt   = rem % NT;
    int koff = rem / NT;
    int n  = lane & 15;
    int hi = lane >> 4;
    unsigned short* dst = swz + (size_t)frag * 512 + lane * 16;
    #pragma unroll
    for (int h = 0; h < 16; ++h) {
        int k = kb * 32 + (h < 8 ? hi * 8 + h : 16 + hi * 8 + (h - 8));
        float v = (k < CIN) ? w[((size_t)koff * CIN + k) * COUT + nt * 16 + n] : 0.f;
        dst[h] = f2bf_bits(v);
    }
}

// ---------------------------------------------------------------------------
// Sparse conv via WMMA.  grid = (ceil(P/(16*WAVES)), K).  Each wave:
//   loads A fragments (16 gathered rows x CIN, bf16) once via float4 loads,
//   loops COUT/16 tiles: one 32B B-fragment load, WMMA, atomic scatter-add.
// ---------------------------------------------------------------------------
template<int CIN, int COUT, int WAVES>
__global__ __launch_bounds__(WAVES * 32)
void sconv_wmma(const float* __restrict__ fin,            // (Nin+1, CIN), row Nin == 0
                const unsigned short* __restrict__ swz,   // pre-swizzled bf16 weights
                const int*   __restrict__ in_idx,         // (K, P)
                const int*   __restrict__ out_idx,        // (K, P)
                float*       __restrict__ acc,            // (Nout+1, COUT), pre-zeroed
                int P, int Nout) {
    constexpr int KB = (CIN + 31) / 32;            // # of K=32 WMMA steps
    constexpr int NT = COUT / 16;
    const int lane = threadIdx.x & 31;
    const int wave = threadIdx.x >> 5;
    const int tile = blockIdx.x * WAVES + wave;
    const int koff = blockIdx.y;
    if (tile * 16 >= P) return;                    // wave-uniform exit (EXEC stays full)

    const int* ii = in_idx  + (size_t)koff * P;
    const int* oo = out_idx + (size_t)koff * P;

    const int m   = lane & 15;                     // A row / B col / D col
    const int hi  = lane >> 4;                     // lane half
    const int p   = tile * 16 + m;
    const bool pv = (p < P);
    const int row = pv ? ii[p] : 0;                // sentinel rows read the zero row
    const float* frow = fin + (size_t)row * CIN;

    // ---- A fragments: ISA 16-bit A layout: k = (h<8?h:h+8) + hi*8 ----
    BFrag a[KB];
    #pragma unroll
    for (int kb = 0; kb < KB; ++kb) {
        if constexpr (CIN >= 16) {
            // 8-element k-groups are uniformly valid; rows are 32B-aligned.
            #pragma unroll
            for (int g = 0; g < 2; ++g) {              // g=0 -> halves 0..7, g=1 -> 8..15
                int start = kb * 32 + g * 16 + hi * 8;
                float va[8];
                if (pv && (start + 8 <= CIN)) {
                    float4 q0 = *(const float4*)(frow + start);
                    float4 q1 = *(const float4*)(frow + start + 4);
                    va[0] = q0.x; va[1] = q0.y; va[2] = q0.z; va[3] = q0.w;
                    va[4] = q1.x; va[5] = q1.y; va[6] = q1.z; va[7] = q1.w;
                } else {
                    #pragma unroll
                    for (int h = 0; h < 8; ++h) va[h] = 0.f;
                }
                #pragma unroll
                for (int h = 0; h < 8; ++h) a[kb].u[g * 8 + h] = f2bf_bits(va[h]);
            }
        } else {
            #pragma unroll
            for (int h = 0; h < 8; ++h) {
                int kLo = kb * 32 + hi * 8 + h;
                int kHi = kb * 32 + 16 + hi * 8 + h;
                float vLo = (pv && kLo < CIN) ? frow[kLo] : 0.f;
                float vHi = (pv && kHi < CIN) ? frow[kHi] : 0.f;
                a[kb].u[h]     = f2bf_bits(vLo);
                a[kb].u[h + 8] = f2bf_bits(vHi);
            }
        }
    }

    const v16us* wf = (const v16us*)swz;           // 32B-aligned fragments

    #pragma unroll
    for (int nt = 0; nt < NT; ++nt) {
        v8f c = {0.f, 0.f, 0.f, 0.f, 0.f, 0.f, 0.f, 0.f};
        #pragma unroll
        for (int kb = 0; kb < KB; ++kb) {
            BFrag bf;
            bf.u = wf[((size_t)(koff * NT + nt) * KB + kb) * 32 + lane];
            c = __builtin_amdgcn_wmma_f32_16x16x32_bf16(
                    false, a[kb].b, false, bf.b, (short)0, c, false, false);
        }
        // ---- scatter-add: D element (M = r + hi*8, N = lane&15) ----
        #pragma unroll
        for (int r = 0; r < 8; ++r) {
            int mm = r + hi * 8;
            int pp = tile * 16 + mm;
            if (pp < P) {
                int o = oo[pp];
                if (o < Nout)
                    atomicAdd(&acc[(size_t)o * COUT + nt * 16 + m], c[r]);
            }
        }
    }
}

// ---------------------------------------------------------------------------
// Fused eval-BatchNorm + ReLU; also writes the zero sentinel row at Nout.
// bn layout: (4, C) = gamma | beta | mean | var
// ---------------------------------------------------------------------------
__global__ void bnrelu_kernel(const float* __restrict__ acc,
                              const float* __restrict__ bn,
                              float* __restrict__ out, int Nout, int C) {
    int idx = blockIdx.x * blockDim.x + threadIdx.x;
    int total = (Nout + 1) * C;
    if (idx >= total) return;
    int i = idx / C, c = idx - i * C;
    float y = 0.f;
    if (i < Nout) {
        float g  = bn[c];
        float b  = bn[C + c];
        float mu = bn[2 * C + c];
        float vr = bn[3 * C + c];
        y = fmaxf((acc[idx] - mu) * rsqrtf(vr + 1e-3f) * g + b, 0.f);
    }
    out[idx] = y;
}

// ---------------------------------------------------------------------------
// Per-batch global max over active voxels. Values are post-ReLU (>= 0) so a
// uint atomicMax on the float bit pattern is exact; d_out pre-zeroed.
// ---------------------------------------------------------------------------
__global__ void segmax_kernel(const float* __restrict__ f,
                              const int* __restrict__ bidx,
                              float* __restrict__ out, int n5) {
    int idx = blockIdx.x * blockDim.x + threadIdx.x;
    if (idx >= n5 * 128) return;
    int i = idx >> 7, c = idx & 127;
    float v = fmaxf(f[idx], 0.f);
    atomicMax((unsigned int*)&out[bidx[i] * 128 + c], __float_as_uint(v));
}

// ---------------------------------------------------------------------------
// Host-side layer driver: pre-swizzle weights, zero acc, conv, BN+ReLU.
// ---------------------------------------------------------------------------
static void run_conv(const float* fin, const float* w, const float* bn,
                     const int* ii, const int* oo, unsigned short* swz,
                     float* acc, float* fout,
                     int K, int P, int Cin, int Cout, int Nout, hipStream_t s) {
    (void)hipMemsetAsync(acc, 0, sizeof(float) * (size_t)(Nout + 1) * Cout, s);
    constexpr int WAVES = 4;
    const int KB = (Cin + 31) / 32;
    const int NT = Cout / 16;
    int swzThreads = K * NT * KB * 32;
    dim3 pgrid((swzThreads + 255) / 256), pblk(256);
    dim3 grid((P + 16 * WAVES - 1) / (16 * WAVES), K, 1);
    dim3 blk(32 * WAVES);
    if (Cin == 4 && Cout == 16) {
        preswz_kernel< 4, 16><<<pgrid, pblk, 0, s>>>(w, swz, K);
        sconv_wmma< 4, 16, WAVES><<<grid, blk, 0, s>>>(fin, swz, ii, oo, acc, P, Nout);
    } else if (Cin == 16 && Cout == 16) {
        preswz_kernel<16, 16><<<pgrid, pblk, 0, s>>>(w, swz, K);
        sconv_wmma<16, 16, WAVES><<<grid, blk, 0, s>>>(fin, swz, ii, oo, acc, P, Nout);
    } else if (Cin == 16 && Cout == 32) {
        preswz_kernel<16, 32><<<pgrid, pblk, 0, s>>>(w, swz, K);
        sconv_wmma<16, 32, WAVES><<<grid, blk, 0, s>>>(fin, swz, ii, oo, acc, P, Nout);
    } else if (Cin == 32 && Cout == 32) {
        preswz_kernel<32, 32><<<pgrid, pblk, 0, s>>>(w, swz, K);
        sconv_wmma<32, 32, WAVES><<<grid, blk, 0, s>>>(fin, swz, ii, oo, acc, P, Nout);
    } else if (Cin == 32 && Cout == 64) {
        preswz_kernel<32, 64><<<pgrid, pblk, 0, s>>>(w, swz, K);
        sconv_wmma<32, 64, WAVES><<<grid, blk, 0, s>>>(fin, swz, ii, oo, acc, P, Nout);
    } else if (Cin == 64 && Cout == 64) {
        preswz_kernel<64, 64><<<pgrid, pblk, 0, s>>>(w, swz, K);
        sconv_wmma<64, 64, WAVES><<<grid, blk, 0, s>>>(fin, swz, ii, oo, acc, P, Nout);
    } else if (Cin == 64 && Cout == 128) {
        preswz_kernel<64, 128><<<pgrid, pblk, 0, s>>>(w, swz, K);
        sconv_wmma<64, 128, WAVES><<<grid, blk, 0, s>>>(fin, swz, ii, oo, acc, P, Nout);
    }
    int tot = (Nout + 1) * Cout;
    bnrelu_kernel<<<(tot + 255) / 256, 256, 0, s>>>(acc, bn, fout, Nout, Cout);
}

extern "C" void kernel_launch(void* const* d_in, const int* in_sizes, int n_in,
                              void* d_out, int out_size, void* d_ws, size_t ws_size,
                              hipStream_t stream) {
    // dict order: (w0,bn0,...,w11,bn11)=0..23, voxels=24, voxel_num_points=25,
    // subm1_in/out=26/27, down2_in/out=28/29, subm2_in/out=30/31,
    // down3_in/out=32/33, subm3_in/out=34/35, down4_in/out=36/37,
    // subm4_in/out=38/39, out_in/out=40/41, batch_idx=42, n2..n_batch=43..47
    const float* w[12];
    const float* bn[12];
    for (int i = 0; i < 12; ++i) {
        w[i]  = (const float*)d_in[2 * i];
        bn[i] = (const float*)d_in[2 * i + 1];
    }
    const float* voxels = (const float*)d_in[24];
    const int*   vnum   = (const int*)d_in[25];
    const int* subm1_i = (const int*)d_in[26]; const int* subm1_o = (const int*)d_in[27];
    const int* down2_i = (const int*)d_in[28]; const int* down2_o = (const int*)d_in[29];
    const int* subm2_i = (const int*)d_in[30]; const int* subm2_o = (const int*)d_in[31];
    const int* down3_i = (const int*)d_in[32]; const int* down3_o = (const int*)d_in[33];
    const int* subm3_i = (const int*)d_in[34]; const int* subm3_o = (const int*)d_in[35];
    const int* down4_i = (const int*)d_in[36]; const int* down4_o = (const int*)d_in[37];
    const int* subm4_i = (const int*)d_in[38]; const int* subm4_o = (const int*)d_in[39];
    const int* out_i   = (const int*)d_in[40]; const int* out_o   = (const int*)d_in[41];
    const int* batch_idx = (const int*)d_in[42];

    // Exact sizes from in_sizes (subm rulebook padded P == n at that stage,
    // since the (0,0,0) offset pairs every site with itself).
    const int N1  = in_sizes[25];
    const int P1  = in_sizes[26] / 27;
    const int Pd2 = in_sizes[28] / 27;
    const int n2  = in_sizes[30] / 27;
    const int Pd3 = in_sizes[32] / 27;
    const int n3  = in_sizes[34] / 27;
    const int Pd4 = in_sizes[36] / 27;
    const int n4  = in_sizes[38] / 27;
    const int Po  = in_sizes[40] / 3;
    const int n5  = in_sizes[42];          // len(batch_idx)

    // ---- carve workspace (floats, 32B-aligned); feature buffers have +1 zero row
    float* ws = (float*)d_ws;
    size_t off = 0;
    auto carve = [&](size_t n) {
        float* p = ws + off;
        off += (n + 7) & ~(size_t)7;       // keep 32B alignment for fragment loads
        return p;
    };
    float* x0  = carve((size_t)(N1 + 1) * 4);
    float* t1a = carve((size_t)(N1 + 1) * 16);
    float* t1b = carve((size_t)(N1 + 1) * 16);
    float* t2a = carve((size_t)(n2 + 1) * 32);
    float* t2b = carve((size_t)(n2 + 1) * 32);
    float* t3a = carve((size_t)(n3 + 1) * 64);
    float* t3b = carve((size_t)(n3 + 1) * 64);
    float* t4a = carve((size_t)(n4 + 1) * 64);
    float* t4b = carve((size_t)(n4 + 1) * 64);
    float* t5  = carve((size_t)(n5 + 1) * 128);
    // bf16 weight-fragment buffer: worst case 27 offsets * KB=2 * NT=8 frags
    unsigned short* swz = (unsigned short*)carve((size_t)27 * 2 * 8 * 512 / 2);
    size_t accmax = (size_t)(N1 + 1) * 16;
    if ((size_t)(n2 + 1) * 32  > accmax) accmax = (size_t)(n2 + 1) * 32;
    if ((size_t)(n3 + 1) * 64  > accmax) accmax = (size_t)(n3 + 1) * 64;
    if ((size_t)(n4 + 1) * 64  > accmax) accmax = (size_t)(n4 + 1) * 64;
    if ((size_t)(n5 + 1) * 128 > accmax) accmax = (size_t)(n5 + 1) * 128;
    float* acc = carve(accmax);

    // ---- mean-VFE ----
    {
        int tot = (N1 + 1) * 4;
        vfe_kernel<<<(tot + 255) / 256, 256, 0, stream>>>(voxels, vnum, x0, N1);
    }

    // ---- 12 sparse conv + BN + ReLU layers ----
    run_conv(x0,  w[0],  bn[0],  subm1_i, subm1_o, swz, acc, t1a, 27, P1,   4,  16, N1, stream);
    run_conv(t1a, w[1],  bn[1],  subm1_i, subm1_o, swz, acc, t1b, 27, P1,  16,  16, N1, stream);
    run_conv(t1b, w[2],  bn[2],  down2_i, down2_o, swz, acc, t2a, 27, Pd2, 16,  32, n2, stream);
    run_conv(t2a, w[3],  bn[3],  subm2_i, subm2_o, swz, acc, t2b, 27, n2,  32,  32, n2, stream);
    run_conv(t2b, w[4],  bn[4],  subm2_i, subm2_o, swz, acc, t2a, 27, n2,  32,  32, n2, stream);
    run_conv(t2a, w[5],  bn[5],  down3_i, down3_o, swz, acc, t3a, 27, Pd3, 32,  64, n3, stream);
    run_conv(t3a, w[6],  bn[6],  subm3_i, subm3_o, swz, acc, t3b, 27, n3,  64,  64, n3, stream);
    run_conv(t3b, w[7],  bn[7],  subm3_i, subm3_o, swz, acc, t3a, 27, n3,  64,  64, n3, stream);
    run_conv(t3a, w[8],  bn[8],  down4_i, down4_o, swz, acc, t4a, 27, Pd4, 64,  64, n4, stream);
    run_conv(t4a, w[9],  bn[9],  subm4_i, subm4_o, swz, acc, t4b, 27, n4,  64,  64, n4, stream);
    run_conv(t4b, w[10], bn[10], subm4_i, subm4_o, swz, acc, t4a, 27, n4,  64,  64, n4, stream);
    run_conv(t4a, w[11], bn[11], out_i,   out_o,   swz, acc, t5,   3, Po,  64, 128, n5, stream);

    // ---- per-batch segment max ----
    (void)hipMemsetAsync(d_out, 0, sizeof(float) * (size_t)out_size, stream);
    {
        int tot = n5 * 128;
        segmax_kernel<<<(tot + 255) / 256, 256, 0, stream>>>(t5, batch_idx, (float*)d_out, n5);
    }
}
